// SelfAttentionSparse_34230889349470
// MI455X (gfx1250) — matrix operand order repaired and checked by reading, not compiled, not run on hardware
//
#include <hip/hip_runtime.h>

// ---------------------------------------------------------------------------
// SelfAttention (B=4, DIM=512, L=2048, H=8, INNER=EMB=64) for gfx1250 (MI455X)
// All GEMMs via v_wmma_f32_16x16x32_bf16 (wave32). Two-pass softmax (stats
// then fused recompute+O). Scratch ~45MB -> fully L2-resident (192MB L2).
// CDNA5 paths: v_wmma, global_load_async_to_lds_b128 (+s_wait_asynccnt),
// global_prefetch_b8.
// ---------------------------------------------------------------------------

#define B_     4
#define DIM_   512
#define L_     2048
#define H_     8
#define INNER_ 64
#define EMB_   64
#define SCALE_ 0.125f   // 1/sqrt(64)

typedef __attribute__((ext_vector_type(16))) __bf16 v16bf;
typedef __attribute__((ext_vector_type(8)))  float  v8f;

union FragB16 { v16bf v; unsigned short s[16]; unsigned u[8]; };

static __device__ __forceinline__ unsigned short f2bf(float f) {
  unsigned u = __builtin_bit_cast(unsigned, f);
  u += 0x7FFFu + ((u >> 16) & 1u);          // round-to-nearest-even
  return (unsigned short)(u >> 16);
}

static __device__ __forceinline__ v8f v8zero() {
  v8f x;
#pragma unroll
  for (int i = 0; i < 8; ++i) x[i] = 0.0f;
  return x;
}

// K-offset of VGPR pair v for the 16x32 bf16 A fragment (contiguous order):
// lanes 0-15: V0..7 -> K 0..15 ; lanes 16-31: K 16..31
static __device__ __forceinline__ int a_kbase(int v, int hi) {
  return 2 * v + 16 * hi;
}

// Low 32 bits of a flat pointer into LDS = byte offset within the group's LDS
static __device__ __forceinline__ unsigned lds_off32(const void* p) {
  return (unsigned)(reinterpret_cast<unsigned long long>(p));
}

// ---------------------------------------------------------------------------
// Elementwise conversions
// ---------------------------------------------------------------------------
__global__ void k_cvt_bf16(const float* __restrict__ in,
                           unsigned short* __restrict__ out, int n) {
  int i = blockIdx.x * blockDim.x + threadIdx.x;
  if (i < n) out[i] = f2bf(in[i]);
}

// unifyheads [H][DIM][EMB] -> Ubw [DIM][H*EMB]  (row c, col h*64+t)
__global__ void k_pack_unify(const float* __restrict__ U,
                             unsigned short* __restrict__ out) {
  int i = blockIdx.x * blockDim.x + threadIdx.x;
  int n = H_ * DIM_ * EMB_;
  if (i >= n) return;
  int h = i / (DIM_ * EMB_);
  int r = i % (DIM_ * EMB_);
  int c = r / EMB_;
  int t = r % EMB_;
  out[c * (H_ * EMB_) + h * EMB_ + t] = f2bf(U[i]);
}

// ---------------------------------------------------------------------------
// Generic 64xN GEMM tile: C[64 x 16] = A[64 x K] * Bm[K x N] (+bias per row)
// Block = 128 threads (4 waves); wave w computes rows [16w,16w+16).
// B tile staged transposed in LDS (stride 34 ushorts to dodge bank conflicts).
// ---------------------------------------------------------------------------
template <bool OUT_BF16>
static __device__ __forceinline__ void gemm64_tile(
    const unsigned short* __restrict__ A,    // 64+ rows x K, row-major
    const unsigned short* __restrict__ Bm,   // K x N, row-major
    void* __restrict__ Cout,                 // rows x N
    const float* __restrict__ bias,          // 64 floats (per row)
    int K, int N, int n0, unsigned short* ldsB /* [16][34] */) {
  const int tid  = threadIdx.x;
  const int wave = tid >> 5, lane = tid & 31;
  const int r = lane & 15, hi = lane >> 4;
  const int m0 = wave * 16;

  v8f c = v8zero();
  for (int ks = 0; ks < K; ks += 32) {
    __syncthreads();
    // stage Bm[ks..ks+32) x [n0..n0+16) transposed -> ldsB[n][k]
#pragma unroll
    for (int q = 0; q < 2; ++q) {
      int u    = tid * 2 + q;        // 0..255
      int krow = u >> 3;             // 0..31
      int nc   = (u & 7) * 2;        // 0,2,..,14
      const unsigned short* src = Bm + (size_t)(ks + krow) * N + n0 + nc;
      unsigned val = *reinterpret_cast<const unsigned*>(src);
      if (ks + 32 < K)               // warm next K-step (global_prefetch_b8)
        __builtin_prefetch(src + (size_t)32 * N, 0, 1);
      ldsB[(nc + 0) * 34 + krow] = (unsigned short)(val & 0xFFFFu);
      ldsB[(nc + 1) * 34 + krow] = (unsigned short)(val >> 16);
    }
    __syncthreads();

    FragB16 a, b;
#pragma unroll
    for (int v = 0; v < 8; ++v) {
      int kb = a_kbase(v, hi);       // A: packed K pair, K-fastest source
      a.u[v] = *reinterpret_cast<const unsigned*>(
          A + (size_t)(m0 + r) * K + ks + kb);
      int kbb = hi * 16 + 2 * v;     // B frag: lanes 0-15 K 0..15, 16-31 K 16..31
      b.u[v] = *reinterpret_cast<const unsigned*>(ldsB + r * 34 + kbb);
    }
    c = __builtin_amdgcn_wmma_f32_16x16x32_bf16(false, a.v, false, b.v,
                                                (short)0, c, false, false);
  }

#pragma unroll
  for (int v = 0; v < 8; ++v) {
    int m = m0 + v + 8 * hi;
    float val = c[v] + bias[m];
    size_t idx = (size_t)m * N + n0 + r;
    if (OUT_BF16) ((unsigned short*)Cout)[idx] = f2bf(val);
    else          ((float*)Cout)[idx] = val;
  }
}

// ---------------------------------------------------------------------------
// K/Q/V projection: per (b,h,p) C[64 x 2048] = W[64 x 512] * xb[512 x 2048]
// ---------------------------------------------------------------------------
__global__ __launch_bounds__(128) void k_proj(
    const unsigned short* __restrict__ xb,
    const unsigned short* __restrict__ Wk,
    const unsigned short* __restrict__ Wq,
    const unsigned short* __restrict__ Wv,
    const float* __restrict__ bK, const float* __restrict__ bQ,
    const float* __restrict__ bV,
    unsigned short* __restrict__ Kb, unsigned short* __restrict__ Qb,
    unsigned short* __restrict__ Vb) {
  __shared__ unsigned short lds[16 * 34];
  int z = blockIdx.y;                 // 0 .. B*3*H-1
  int b = z / (3 * H_);
  int rm = z % (3 * H_);
  int p = rm / H_;
  int h = rm % H_;
  const unsigned short* W = (p == 0 ? Wk : (p == 1 ? Wq : Wv)) +
                            (size_t)h * INNER_ * DIM_;
  const float* bias = (p == 0 ? bK : (p == 1 ? bQ : bV)) + h * INNER_;
  unsigned short* out = (p == 0 ? Kb : (p == 1 ? Qb : Vb)) +
                        (size_t)(b * H_ + h) * INNER_ * L_;
  const unsigned short* X = xb + (size_t)b * DIM_ * L_;
  gemm64_tile<true>(W, X, out, bias, DIM_, L_, blockIdx.x * 16, lds);
}

// ---------------------------------------------------------------------------
// Pass 1: per-row softmax stats. S[i,j] = scale * sum_e K[e,i] Q[e,j].
// Wave owns a 16-row i tile (A = K^T frags held in regs), streams j tiles;
// Q tile staged once per j-step in LDS (shared by all 8 waves). Online
// max/sum across the 16 lanes holding each row (shfl_xor within halves).
// ---------------------------------------------------------------------------
__global__ __launch_bounds__(256) void k_stats(
    const unsigned short* __restrict__ Kb,
    const unsigned short* __restrict__ Qb,
    float* __restrict__ mOut, float* __restrict__ zOut) {
  __shared__ unsigned short ldsQ[16 * 66];
  const int bh   = blockIdx.y;
  const int tid  = threadIdx.x;
  const int wave = tid >> 5, lane = tid & 31;
  const int r = lane & 15, hi = lane >> 4;
  const int i0 = (blockIdx.x * 8 + wave) * 16;
  const unsigned short* Kp = Kb + (size_t)bh * INNER_ * L_;
  const unsigned short* Qp = Qb + (size_t)bh * INNER_ * L_;

  // A = K^T tile [m=i, k=e], e=0..63 -> two fragments (strided gathers, once)
  FragB16 a[2];
#pragma unroll
  for (int ek = 0; ek < 2; ++ek)
#pragma unroll
    for (int v = 0; v < 8; ++v) {
      int kb = a_kbase(v, hi) + 32 * ek;
      a[ek].s[2 * v + 0] = Kp[(size_t)(kb + 0) * L_ + i0 + r];
      a[ek].s[2 * v + 1] = Kp[(size_t)(kb + 1) * L_ + i0 + r];
    }

  float mrun[8], zrun[8];
#pragma unroll
  for (int v = 0; v < 8; ++v) { mrun[v] = -3.0e38f; zrun[v] = 0.0f; }

  for (int j0 = 0; j0 < L_; j0 += 16) {
    __syncthreads();
    // stage Q[0..64)e x [j0..j0+16) transposed -> ldsQ[j][e]
#pragma unroll
    for (int q = 0; q < 2; ++q) {
      int u  = tid * 2 + q;          // 0..511
      int e  = u >> 3;               // 0..63
      int jc = (u & 7) * 2;
      const unsigned short* src = Qp + (size_t)e * L_ + j0 + jc;
      unsigned val = *reinterpret_cast<const unsigned*>(src);
      if (j0 + 16 < L_) __builtin_prefetch(src + 16, 0, 1);
      ldsQ[(jc + 0) * 66 + e] = (unsigned short)(val & 0xFFFFu);
      ldsQ[(jc + 1) * 66 + e] = (unsigned short)(val >> 16);
    }
    __syncthreads();

    v8f c = v8zero();
#pragma unroll
    for (int ek = 0; ek < 2; ++ek) {
      FragB16 bq;
#pragma unroll
      for (int v = 0; v < 8; ++v) {
        int kbb = 32 * ek + hi * 16 + 2 * v;
        bq.u[v] = *reinterpret_cast<const unsigned*>(ldsQ + r * 66 + kbb);
      }
      c = __builtin_amdgcn_wmma_f32_16x16x32_bf16(false, a[ek].v, false, bq.v,
                                                  (short)0, c, false, false);
    }
#pragma unroll
    for (int v = 0; v < 8; ++v) {
      float s = c[v] * SCALE_;
      float tmax = s;
#pragma unroll
      for (int off = 1; off < 16; off <<= 1)
        tmax = fmaxf(tmax, __shfl_xor(tmax, off, 32));
      float nm = fmaxf(mrun[v], tmax);
      float p = __expf(s - nm);
      float ps = p;
#pragma unroll
      for (int off = 1; off < 16; off <<= 1) ps += __shfl_xor(ps, off, 32);
      zrun[v] = zrun[v] * __expf(mrun[v] - nm) + ps;
      mrun[v] = nm;
    }
  }
  if (r == 0) {
#pragma unroll
    for (int v = 0; v < 8; ++v) {
      int row = i0 + v + 8 * hi;
      mOut[(size_t)bh * L_ + row] = mrun[v];
      zOut[(size_t)bh * L_ + row] = zrun[v];
    }
  }
}

// ---------------------------------------------------------------------------
// Pass 2 (fused): recompute S^T tiles, E' = exp(scale*S - m_i)/Z_i, and
// accumulate O[t,j] = sum_i V[t,i] * E'[i,j]. Wave owns a 16-wide j tile and
// 4 accumulator tiles over t. K tile (64e x 32i, row-major, 64B rows) is
// staged each step with GLOBAL_LOAD_ASYNC_TO_LDS_B128 (ASYNCcnt), then
// gathered per-lane as B fragments; E converted C-layout -> B-fragment via a
// per-wave LDS round trip.
// ---------------------------------------------------------------------------
__global__ __launch_bounds__(128) void k_attn(
    const unsigned short* __restrict__ Kb,
    const unsigned short* __restrict__ Qb,
    const unsigned short* __restrict__ Vb,
    const float* __restrict__ mIn, const float* __restrict__ zIn,
    unsigned short* __restrict__ Ob) {
  __shared__ __align__(16) unsigned short ldsK[64 * 32];  // [e][i], 64B rows
  __shared__ unsigned short ldsE[4][16 * 34];
  const int bh   = blockIdx.y;
  const int tid  = threadIdx.x;
  const int wave = tid >> 5, lane = tid & 31;
  const int r = lane & 15, hi = lane >> 4;
  const int j0 = (blockIdx.x * 4 + wave) * 16;
  const unsigned short* Kp = Kb + (size_t)bh * INNER_ * L_;
  const unsigned short* Qp = Qb + (size_t)bh * INNER_ * L_;
  const unsigned short* Vp = Vb + (size_t)bh * INNER_ * L_;
  const float* mp = mIn + (size_t)bh * L_;
  const float* zp = zIn + (size_t)bh * L_;
  const unsigned ldsKBase = lds_off32(ldsK);

  // A = Q^T tile [m=j, k=e] (fixed, loaded once)
  FragB16 aq[2];
#pragma unroll
  for (int ek = 0; ek < 2; ++ek)
#pragma unroll
    for (int v = 0; v < 8; ++v) {
      int kb = a_kbase(v, hi) + 32 * ek;
      aq[ek].s[2 * v + 0] = Qp[(size_t)(kb + 0) * L_ + j0 + r];
      aq[ek].s[2 * v + 1] = Qp[(size_t)(kb + 1) * L_ + j0 + r];
    }

  v8f acc[4];
#pragma unroll
  for (int t = 0; t < 4; ++t) acc[t] = v8zero();

  for (int i0 = 0; i0 < L_; i0 += 32) {
    __syncthreads();  // previous step's ldsK reads complete in all waves
    // DMA K[0..64)e x [i0..i0+32) into LDS: 64 rows x 64B = 4KB.
    // 128 lanes x 16B per async-B128 -> 2 passes, no VGPR round trip.
#pragma unroll
    for (int q = 0; q < 2; ++q) {
      int u   = tid + q * 128;       // 0..255
      int e   = u >> 2;              // 0..63
      int seg = (u & 3) * 16;        // byte segment within 64B row
      const char* gsrc =
          (const char*)(Kp + (size_t)e * L_ + i0) + seg;
      unsigned ldst = ldsKBase + (unsigned)(e * 64 + seg);
      asm volatile("global_load_async_to_lds_b128 %0, %1, off"
                   :: "v"(ldst), "v"(gsrc) : "memory");
      if (i0 + 32 < L_) __builtin_prefetch(gsrc + 64, 0, 1);  // next i-step
    }
    asm volatile("s_wait_asynccnt 0x0" ::: "memory");
    __syncthreads();

    // two S^T tiles (16j x 16i each) -> E' into this wave's LDS_E [j][i]
#pragma unroll
    for (int it = 0; it < 2; ++it) {
      const int iLoc = 16 * it + r;
      v8f c = v8zero();
#pragma unroll
      for (int ek = 0; ek < 2; ++ek) {
        FragB16 bk;  // B = K tile [k=e, n=i]: lane n -> column i, e strided
#pragma unroll
        for (int v = 0; v < 8; ++v) {
          int e0 = 32 * ek + hi * 16 + 2 * v;
          bk.s[2 * v + 0] = ldsK[(e0 + 0) * 32 + iLoc];
          bk.s[2 * v + 1] = ldsK[(e0 + 1) * 32 + iLoc];
        }
        c = __builtin_amdgcn_wmma_f32_16x16x32_bf16(false, aq[ek].v, false,
                                                    bk.v, (short)0, c, false,
                                                    false);
      }
      int   ig   = i0 + iLoc;          // column index i (same for all v)
      float mi   = mp[ig];
      float zinv = 1.0f / zp[ig];
#pragma unroll
      for (int v = 0; v < 8; ++v) {
        float e = __expf(c[v] * SCALE_ - mi) * zinv;
        ldsE[wave][(v + 8 * hi) * 34 + iLoc] = f2bf(e);
      }
    }
    __syncthreads();   // E visible; also fences ldsK for next iteration

    // B = E' [k=i(32), n=j(16)]
    FragB16 be;
#pragma unroll
    for (int v = 0; v < 8; ++v) {
      int kbb = hi * 16 + 2 * v;
      be.u[v] = *reinterpret_cast<const unsigned*>(&ldsE[wave][r * 34 + kbb]);
    }
    // O tiles: A = V [m=t, k=i] (K-fastest source -> packed pair loads)
#pragma unroll
    for (int t = 0; t < 4; ++t) {
      FragB16 av;
#pragma unroll
      for (int v = 0; v < 8; ++v) {
        int kb = a_kbase(v, hi);
        av.u[v] = *reinterpret_cast<const unsigned*>(
            Vp + (size_t)(16 * t + r) * L_ + i0 + kb);
      }
      acc[t] = __builtin_amdgcn_wmma_f32_16x16x32_bf16(false, av.v, false,
                                                       be.v, (short)0, acc[t],
                                                       false, false);
    }
  }

  // store O as bf16, rows (h*64 + t) within batch b: global row = bh*64 + ...
#pragma unroll
  for (int t = 0; t < 4; ++t)
#pragma unroll
    for (int v = 0; v < 8; ++v) {
      int row = bh * 64 + 16 * t + v + 8 * hi;
      Ob[(size_t)row * L_ + j0 + r] = f2bf(acc[t][v]);
    }
}

// ---------------------------------------------------------------------------
// Final: out[b] = Ubw[512x512] * O_cat[512x2048] + bias (f32 output)
// ---------------------------------------------------------------------------
__global__ __launch_bounds__(128) void k_final(
    const unsigned short* __restrict__ Ubw,
    const unsigned short* __restrict__ Ob,
    const float* __restrict__ ubias, float* __restrict__ out) {
  __shared__ unsigned short lds[16 * 34];
  int b = blockIdx.z, mg = blockIdx.y;
  gemm64_tile<false>(Ubw + (size_t)mg * 64 * (H_ * EMB_),
                     Ob + (size_t)b * (H_ * EMB_) * L_,
                     out + (size_t)b * DIM_ * L_ + (size_t)mg * 64 * L_,
                     ubias + mg * 64, H_ * EMB_, L_, blockIdx.x * 16, lds);
}

// ---------------------------------------------------------------------------
extern "C" void kernel_launch(void* const* d_in, const int* in_sizes, int n_in,
                              void* d_out, int out_size, void* d_ws,
                              size_t ws_size, hipStream_t stream) {
  const float* x     = (const float*)d_in[0];
  const float* tokey = (const float*)d_in[1];
  const float* kbias = (const float*)d_in[2];
  const float* toqry = (const float*)d_in[3];
  const float* qbias = (const float*)d_in[4];
  const float* toval = (const float*)d_in[5];
  const float* vbias = (const float*)d_in[6];
  const float* unify = (const float*)d_in[7];
  const float* ubias = (const float*)d_in[8];
  float* out = (float*)d_out;

  const size_t nX  = (size_t)B_ * DIM_ * L_;          // 4.19M
  const size_t nW  = (size_t)H_ * INNER_ * DIM_;      // 262k per projection
  const size_t nU  = (size_t)DIM_ * H_ * EMB_;        // 262k
  const size_t nKQV= (size_t)B_ * H_ * INNER_ * L_;   // 4.19M
  const size_t nMZ = (size_t)B_ * H_ * L_;            // 65k

  char* ws = (char*)d_ws;
  size_t off = 0;
  auto alloc = [&](size_t bytes) {
    size_t o = off;
    off += (bytes + 255) & ~(size_t)255;
    return o;
  };
  unsigned short* xb  = (unsigned short*)(ws + alloc(nX * 2));
  unsigned short* Wk  = (unsigned short*)(ws + alloc(nW * 2));
  unsigned short* Wq  = (unsigned short*)(ws + alloc(nW * 2));
  unsigned short* Wv  = (unsigned short*)(ws + alloc(nW * 2));
  unsigned short* Ubw = (unsigned short*)(ws + alloc(nU * 2));
  unsigned short* Kb  = (unsigned short*)(ws + alloc(nKQV * 2));
  unsigned short* Qb  = (unsigned short*)(ws + alloc(nKQV * 2));
  unsigned short* Vb  = (unsigned short*)(ws + alloc(nKQV * 2));
  float*          mB  = (float*)(ws + alloc(nMZ * 4));
  float*          zB  = (float*)(ws + alloc(nMZ * 4));
  unsigned short* Ob  = (unsigned short*)(ws + alloc(nX * 2));
  (void)ws_size; (void)in_sizes; (void)n_in; (void)out_size;

  // 0) bf16 conversions + weight packing
  k_cvt_bf16<<<dim3((unsigned)((nX + 255) / 256)), 256, 0, stream>>>(x, xb, (int)nX);
  k_cvt_bf16<<<dim3((unsigned)((nW + 255) / 256)), 256, 0, stream>>>(tokey, Wk, (int)nW);
  k_cvt_bf16<<<dim3((unsigned)((nW + 255) / 256)), 256, 0, stream>>>(toqry, Wq, (int)nW);
  k_cvt_bf16<<<dim3((unsigned)((nW + 255) / 256)), 256, 0, stream>>>(toval, Wv, (int)nW);
  k_pack_unify<<<dim3((unsigned)((nU + 255) / 256)), 256, 0, stream>>>(unify, Ubw);

  // 1) K/Q/V projections (WMMA GEMMs)
  k_proj<<<dim3(L_ / 16, B_ * 3 * H_), 128, 0, stream>>>(
      xb, Wk, Wq, Wv, kbias, qbias, vbias, Kb, Qb, Vb);

  // 2) softmax row stats (WMMA + online max/sum)
  k_stats<<<dim3(L_ / 128, B_ * H_), 256, 0, stream>>>(Kb, Qb, mB, zB);

  // 3) fused S-recompute + exp + O = V * E'  (async LDS staging)
  k_attn<<<dim3(L_ / 64, B_ * H_), 128, 0, stream>>>(Kb, Qb, Vb, mB, zB, Ob);

  // 4) head unification (+bias), f32 output
  k_final<<<dim3(L_ / 16, DIM_ / 64, B_), 128, 0, stream>>>(Ubw, Ob, ubias, out);
}
